// ElmanRNN_58145267254150
// MI455X (gfx1250) — matrix operand order repaired and checked by reading, not compiled, not run on hardware
//
#include <hip/hip_runtime.h>
#include <hip/hip_bf16.h>
#include <math.h>

// ---------------------------------------------------------------------------
// Elman RNN (2 layers) for MI455X / gfx1250, wave32, f16 WMMA w/ f32 accum.
//   B=64, T=1024, I=H=256
// Pipeline:
//   0) convert all 4 weight matrices fp32 -> f16 in ws
//   1) xp0 = f16( x @ W_ih0^T + b_ih0 )            (WMMA GEMM, streaming)
//   2) scan layer0: h_t = tanh(xp0_t + h@W_hh0^T)  (LDS-resident W, 4 WGs)
//        -> hseq f16
//   3) xp1 = f16( hseq @ W_ih1^T + b_ih1 )
//   4) scan layer1 -> d_out fp32
// Scan is latency-bound (2048 serial steps): W_hh lives in LDS, h is
// double-buffered in LDS (1 barrier/step), and the xp fragment is
// register-pipelined one timestep ahead so global latency is fully hidden.
// ---------------------------------------------------------------------------

typedef __attribute__((ext_vector_type(16))) _Float16 v16h;
typedef __attribute__((ext_vector_type(8)))  _Float16 v8h;
typedef __attribute__((ext_vector_type(8)))  float    v8f;

#define RNN_B 64
#define RNN_T 1024
#define RNN_H 256          // == I
#define LDS_PITCH 264      // 256 + 8 halfs: row stride = 132 dwords ≡ 4 (mod 64 banks)

static __device__ __forceinline__ v16h cat16(v8h lo, v8h hi) {
  return __builtin_shufflevector(lo, hi, 0,1,2,3,4,5,6,7,8,9,10,11,12,13,14,15);
}

static __device__ __forceinline__ v8f wmma_f16(v16h a, v16h b, v8f c) {
  // D = A(16x32,f16) * B(32x16,f16) + C(16x16,f32)
  return __builtin_amdgcn_wmma_f32_16x16x32_f16(
      /*neg_a=*/false, a, /*neg_b=*/false, b,
      /*c_mod=*/(short)0, c, /*reuse_a=*/false, /*reuse_b=*/false);
}

// ---------------------------------------------------------------------------
// Kernel 0: fp32 -> f16 weight conversion (4 matrices of 256x256, contiguous)
// ---------------------------------------------------------------------------
__global__ void cvt_weights_kernel(const float* __restrict__ w0,
                                   const float* __restrict__ w1,
                                   const float* __restrict__ w2,
                                   const float* __restrict__ w3,
                                   _Float16* __restrict__ dst) {
  int idx = blockIdx.x * 256 + threadIdx.x;        // 0 .. 4*65536-1
  const float* srcs[4] = {w0, w1, w2, w3};
  dst[idx] = (_Float16)srcs[idx >> 16][idx & 65535];
}

// ---------------------------------------------------------------------------
// Kernel 1/3: input projection GEMM   xp[m,n] = f16( sum_k A[m,k]*W[n,k] + b[n] )
//   M = B*T = 65536, N = K = 256. Block = 256 thr (8 waves), WG tile 32x256.
//   Wave (msub, nsub): 16 rows x 64 cols (4 WMMA accumulators).
// ---------------------------------------------------------------------------
template <bool A_IS_F16>
__global__ void xproj_kernel(const void* __restrict__ Ap,
                             const _Float16* __restrict__ W,   // [n][k] f16
                             const float* __restrict__ bias,
                             _Float16* __restrict__ xp) {
  const int tid  = threadIdx.x;
  const int w    = tid >> 5;
  const int lane = tid & 31;
  const int r    = lane & 15;
  const int hi   = lane >> 4;
  const int msub = w >> 2;           // 0..1
  const int nsub = w & 3;            // 0..3
  const size_t m0   = (size_t)blockIdx.x * 32 + msub * 16;
  const size_t arow = m0 + r;

  v8f acc[4];
#pragma unroll
  for (int tt = 0; tt < 4; ++tt) {
    const float bv = bias[nsub * 64 + tt * 16 + r];
#pragma unroll
    for (int j = 0; j < 8; ++j) acc[tt][j] = bv;
  }

#pragma unroll
  for (int kt = 0; kt < 8; ++kt) {
    const int k0   = kt * 32;
    const int off1 = k0 + hi * 8;        // A half 1: K 0..7 / 8..15
    const int off2 = off1 + 16;          // A half 2: K 16..23 / 24..31
    v16h a;
    if constexpr (A_IS_F16) {
      const _Float16* A = (const _Float16*)Ap;
      a = cat16(*(const v8h*)(A + arow * RNN_H + off1),
                *(const v8h*)(A + arow * RNN_H + off2));
    } else {
      const float* A = (const float*)Ap;
      v8f f1 = *(const v8f*)(A + arow * RNN_H + off1);
      v8f f2 = *(const v8f*)(A + arow * RNN_H + off2);
      v8h l, h;
#pragma unroll
      for (int i = 0; i < 8; ++i) { l[i] = (_Float16)f1[i]; h[i] = (_Float16)f2[i]; }
      a = cat16(l, h);
    }
#pragma unroll
    for (int tt = 0; tt < 4; ++tt) {
      const int n = nsub * 64 + tt * 16 + r;
      const _Float16* wb = W + (size_t)n * RNN_H + k0 + hi * 8; // B: K 0..7,16..23 / 8..15,24..31
      v16h b = cat16(*(const v8h*)wb, *(const v8h*)(wb + 16));
      acc[tt] = wmma_f16(a, b, acc[tt]);
    }
  }

#pragma unroll
  for (int tt = 0; tt < 4; ++tt) {
    const int n = nsub * 64 + tt * 16 + r;
#pragma unroll
    for (int j = 0; j < 8; ++j) {
      const size_t m = m0 + hi * 8 + j;
      xp[m * RNN_H + n] = (_Float16)acc[tt][j];
    }
  }
}

// ---------------------------------------------------------------------------
// Kernel 2/4: sequential scan  h_t = tanh(xp_t + h_{t-1} @ W_hh^T)
//   4 WGs, each owns 16 batch rows. 8 waves x 32 cols. W_hh f16 in LDS
//   (padded pitch), h double-buffered in LDS (one barrier per step), xp
//   fragment register-pipelined one step ahead (loads issued at t, waited
//   at t+1 -> global latency fully overlapped with WMMA work).
// ---------------------------------------------------------------------------
#define SCAN_LDS_W   (256 * LDS_PITCH)               // halfs
#define SCAN_LDS_H   (16 * LDS_PITCH)                // halfs per h buffer
#define SCAN_LDS_BYTES ((SCAN_LDS_W + 2 * SCAN_LDS_H) * 2)

__global__ void rnn_scan_kernel(const _Float16* __restrict__ xp,   // [B,T,H] f16
                                const _Float16* __restrict__ Whh,  // [n][k] f16
                                _Float16* __restrict__ hseq,       // [B,T,H] f16 or null
                                float* __restrict__ out32) {       // [B,T,H] f32 or null
  extern __shared__ char smem[];
  _Float16* Wl  = (_Float16*)smem;
  _Float16* hb0 = Wl + SCAN_LDS_W;
  _Float16* hb1 = hb0 + SCAN_LDS_H;

  const int tid  = threadIdx.x;
  const int w    = tid >> 5;
  const int lane = tid & 31;
  const int r    = lane & 15;
  const int hi   = lane >> 4;
  const int n0   = w * 32;
  const int nA   = n0 + r;
  const int nB   = n0 + 16 + r;
  const size_t b0 = (size_t)blockIdx.x * 16;

  // Stage W_hh into LDS with padded pitch (16B chunks, conflict-free).
  for (int c = tid; c < (256 * 256) / 8; c += 256) {
    const int row = c >> 5, c8 = c & 31;
    *(v8h*)(Wl + row * LDS_PITCH + c8 * 8) =
        *(const v8h*)(Whh + row * RNN_H + c8 * 8);
  }
  // h_0 = 0
  for (int i = tid; i < SCAN_LDS_H; i += 256) hb0[i] = (_Float16)0.0f;

  // Register-pipelined xp fragments (double buffered; parity index becomes
  // a compile-time constant under the unroll-2 time loop -> stays in VGPRs).
  _Float16 xA[2][8], xB[2][8];
#pragma unroll
  for (int j = 0; j < 8; ++j) {
    const size_t gb = ((b0 + hi * 8 + j) * RNN_T + 0) * RNN_H;
    xA[0][j] = xp[gb + nA];
    xB[0][j] = xp[gb + nB];
  }
  __syncthreads();

#pragma unroll 2
  for (int t = 0; t < RNN_T; ++t) {
    const int cur = t & 1, nxt = cur ^ 1;
    const _Float16* rd = (t & 1) ? hb1 : hb0;
    _Float16*       wr = (t & 1) ? hb0 : hb1;

    // Issue next timestep's xp loads now; they are waited on at t+1.
    if (t + 1 < RNN_T) {
#pragma unroll
      for (int j = 0; j < 8; ++j) {
        const size_t gb = ((b0 + hi * 8 + j) * RNN_T + (t + 1)) * RNN_H;
        xA[nxt][j] = xp[gb + nA];
        xB[nxt][j] = xp[gb + nB];
      }
    }

    // C init = current xp fragment (f16 -> f32), loaded one step ago.
    v8f c0, c1;
#pragma unroll
    for (int j = 0; j < 8; ++j) {
      c0[j] = (float)xA[cur][j];
      c1[j] = (float)xB[cur][j];
    }

#pragma unroll
    for (int kt = 0; kt < 8; ++kt) {
      const int k0 = kt * 32;
      v16h a = cat16(*(const v8h*)(rd + r * LDS_PITCH + k0 + hi * 8),
                     *(const v8h*)(rd + r * LDS_PITCH + k0 + 16 + hi * 8));
      const _Float16* wbA = Wl + (size_t)nA * LDS_PITCH + k0 + hi * 8;
      v16h bA = cat16(*(const v8h*)wbA, *(const v8h*)(wbA + 16));
      c0 = wmma_f16(a, bA, c0);
      const _Float16* wbB = Wl + (size_t)nB * LDS_PITCH + k0 + hi * 8;
      v16h bB = cat16(*(const v8h*)wbB, *(const v8h*)(wbB + 16));
      c1 = wmma_f16(a, bB, c1);
    }

    // tanh in f32, write back (LDS next-h + global sequence).
#pragma unroll
    for (int j = 0; j < 8; ++j) {
      const float f0 = tanhf(c0[j]);
      const float f1 = tanhf(c1[j]);
      const int   m  = hi * 8 + j;
      wr[m * LDS_PITCH + nA] = (_Float16)f0;
      wr[m * LDS_PITCH + nB] = (_Float16)f1;
      const size_t gb = ((b0 + m) * RNN_T + t) * RNN_H;
      if (hseq)  { hseq[gb + nA] = (_Float16)f0; hseq[gb + nB] = (_Float16)f1; }
      if (out32) { out32[gb + nA] = f0;          out32[gb + nB] = f1; }
    }
    __syncthreads();
  }
}

// ---------------------------------------------------------------------------
// Host launcher. Inputs: x, W_ih0, b_ih0, W_hh0, W_ih1, b_ih1, W_hh1.
// ws layout: [w16: 4*65536 f16][xp: B*T*H f16][hseq: B*T*H f16]  (~67.6 MB)
// ---------------------------------------------------------------------------
extern "C" void kernel_launch(void* const* d_in, const int* in_sizes, int n_in,
                              void* d_out, int out_size, void* d_ws, size_t ws_size,
                              hipStream_t stream) {
  (void)in_sizes; (void)n_in; (void)out_size; (void)ws_size;
  const float* x     = (const float*)d_in[0];
  const float* W_ih0 = (const float*)d_in[1];
  const float* b_ih0 = (const float*)d_in[2];
  const float* W_hh0 = (const float*)d_in[3];
  const float* W_ih1 = (const float*)d_in[4];
  const float* b_ih1 = (const float*)d_in[5];
  const float* W_hh1 = (const float*)d_in[6];
  float* out = (float*)d_out;

  const size_t WELE = 256 * 256;                 // 65536 halfs per matrix
  _Float16* w16  = (_Float16*)d_ws;
  _Float16* xpb  = (_Float16*)((char*)d_ws + 4 * WELE * sizeof(_Float16));
  _Float16* hseq = xpb + (size_t)RNN_B * RNN_T * RNN_H;

  (void)hipFuncSetAttribute(reinterpret_cast<const void*>(rnn_scan_kernel),
                            hipFuncAttributeMaxDynamicSharedMemorySize,
                            SCAN_LDS_BYTES);

  // 0) weights -> f16
  cvt_weights_kernel<<<1024, 256, 0, stream>>>(W_ih0, W_hh0, W_ih1, W_hh1, w16);

  // 1) layer0 input projection (A fp32)
  xproj_kernel<false><<<(RNN_B * RNN_T) / 32, 256, 0, stream>>>(
      (const void*)x, w16 + 0 * WELE, b_ih0, xpb);

  // 2) layer0 scan -> hseq (f16)
  rnn_scan_kernel<<<RNN_B / 16, 256, SCAN_LDS_BYTES, stream>>>(
      xpb, w16 + 1 * WELE, hseq, nullptr);

  // 3) layer1 input projection (A f16 = hseq), reuse xp buffer
  xproj_kernel<true><<<(RNN_B * RNN_T) / 32, 256, 0, stream>>>(
      (const void*)hseq, w16 + 2 * WELE, b_ih1, xpb);

  // 4) layer1 scan -> final fp32 output
  rnn_scan_kernel<<<RNN_B / 16, 256, SCAN_LDS_BYTES, stream>>>(
      xpb, w16 + 3 * WELE, nullptr, out);
}
